// Seq2SeqModel_69269232549945
// MI455X (gfx1250) — matrix operand order repaired and checked by reading, not compiled, not run on hardware
//
#include <hip/hip_runtime.h>

typedef __attribute__((ext_vector_type(16))) _Float16 v16h;
typedef __attribute__((ext_vector_type(8)))  float    v8f;

#define V_   27
#define E_   64
#define H_   100
#define S_   10
#define G3P  304    // 3H=300 padded to x16
#define HP   128    // H padded to x32
#define EHP  192    // E+H=164 padded to x32
#define BT   16     // batch rows per workgroup (one WMMA M-tile)
#define NTHR 128    // 4 waves

// ---- WMMA fragment helpers (CDNA5 16x16x32 f16 layouts, ISA 7.12.2) ----

// A fragment: 16x32 tile of row-major f16 matrix p (leading dim ld), rows 0..15, cols k0..k0+31
__device__ __forceinline__ v16h frag_a(const _Float16* p, int ld, int k0, int lane) {
  int m = lane & 15, half = lane >> 4;
  v16h a;
#pragma unroll
  for (int i = 0; i < 16; ++i) {
    int j = i >> 1, kk = i & 1;
    int K = ((j >> 2) << 4) + (half << 3) + ((j & 3) << 1) + kk;
    a[i] = p[m * ld + k0 + K];
  }
  return a;
}

// B fragment from TRANSPOSED weights: w is row-major [Npad][Kpad]; logical B[k][n] = w[n][k]
__device__ __forceinline__ v16h frag_bt(const _Float16* w, int ld, int n0, int k0, int lane) {
  int n = lane & 15, half = lane >> 4;
  v16h b;
#pragma unroll
  for (int i = 0; i < 16; ++i) {
    int j = i >> 1, kk = i & 1;
    int K = ((j >> 2) << 4) + (half << 3) + ((j & 3) << 1) + kk;
    b[i] = w[(n0 + n) * ld + k0 + K];
  }
  return b;
}

__device__ __forceinline__ void frag_store(float* p, int ld, int n0, v8f d, int lane) {
  int n = lane & 15, half = lane >> 4;
#pragma unroll
  for (int j = 0; j < 8; ++j) p[(j + half * 8) * ld + n0 + n] = d[j];
}

__device__ __forceinline__ v8f wmma16(v16h a, v16h b, v8f c) {
  return __builtin_amdgcn_wmma_f32_16x16x32_f16(false, a, false, b, (short)0, c, false, false);
}

// ---- weight prep: f32 -> padded/transposed f16 in workspace ----
#define N0 (304*64)    // enc_Wih  [304][64]
#define N1 (304*128)   // enc_Whh  [304][128]
#define N2 (304*192)   // dec_Wih  [304][192]
#define N3 (304*128)   // dec_Whh  [304][128]
#define N4 (112*128)   // attn_W^T [112][128]  (so q = h @ attn_W via bt-load)
#define N5 (32*128)    // lin_W    [32][128]
#define NWTOT (N0+N1+N2+N3+N4+N5)

__global__ void prep_weights(const float* WihE, const float* WhhE,
                             const float* WihD, const float* WhhD,
                             const float* attnW, const float* linW,
                             _Float16* ws) {
  for (int i = blockIdx.x * blockDim.x + threadIdx.x; i < NWTOT; i += gridDim.x * blockDim.x) {
    float v = 0.f;
    int o = i;
    if (o < N0)             { int n = o / 64,  k = o % 64;  if (n < 300)              v = WihE[n * 64  + k]; }
    else if ((o -= N0) < N1){ int n = o / 128, k = o % 128; if (n < 300 && k < 100)   v = WhhE[n * 100 + k]; }
    else if ((o -= N1) < N2){ int n = o / 192, k = o % 192; if (n < 300 && k < 164)   v = WihD[n * 164 + k]; }
    else if ((o -= N2) < N3){ int n = o / 128, k = o % 128; if (n < 300 && k < 100)   v = WhhD[n * 100 + k]; }
    else if ((o -= N3) < N4){ int n = o / 128, k = o % 128; if (n < 100 && k < 100)   v = attnW[k * 100 + n]; }
    else                    { o -= N4; int n = o / 128, k = o % 128; if (n < 27 && k < 100) v = linW[n * 100 + k]; }
    ws[i] = (_Float16)v;
  }
}

// ---- fused seq2seq: encoder GRU -> attention decoder GRU -> logits ----
__global__ __launch_bounds__(NTHR) void seq2seq_kernel(
    const int* __restrict__ encX, const int* __restrict__ decX,
    const float* __restrict__ emb,
    const float* __restrict__ bihE, const float* __restrict__ bhhE,
    const float* __restrict__ bihD, const float* __restrict__ bhhD,
    const float* __restrict__ linb,
    const _Float16* __restrict__ wWihE, const _Float16* __restrict__ wWhhE,
    const _Float16* __restrict__ wWihD, const _Float16* __restrict__ wWhhD,
    const _Float16* __restrict__ wAttnT, const _Float16* __restrict__ wLin,
    float* __restrict__ out) {
  __shared__ float    sH [BT * HP];       // master hidden (f32), cols >=100 stay 0
  __shared__ _Float16 sHh[BT * HP];       // f16 mirror for WMMA A
  __shared__ _Float16 sX [BT * EHP];      // [x | ctx | 0-pad] f16 A matrix
  __shared__ float    sGx[BT * G3P];      // x-gates
  __shared__ float    sGh[BT * G3P];      // h-gates
  __shared__ _Float16 sEnc[S_ * BT * H_]; // encoder outputs
  __shared__ float    sQ [BT * 112];      // attention query q = h @ attn_W
  __shared__ float    sSc[BT * S_];       // scores / softmax weights

  const int tid  = threadIdx.x;
  const int lane = tid & 31;
  const int wv   = tid >> 5;
  const int b0   = blockIdx.x * BT;

  // zero h (incl. K-pad) and the static K-pad of sX (cols 164..191)
  for (int i = tid; i < BT * HP; i += NTHR) { sH[i] = 0.f; sHh[i] = (_Float16)0.f; }
  for (int i = tid; i < BT * (EHP - 164); i += NTHR) {
    int r = i / 28, c = i % 28;
    sX[r * EHP + 164 + c] = (_Float16)0.f;
  }

  // ================= encoder =================
  for (int t = 0; t < S_; ++t) {
    for (int i = tid; i < BT * E_; i += NTHR) {         // gather embedding
      int r = i >> 6, c = i & 63;
      int idx = encX[(b0 + r) * S_ + t];
      sX[r * EHP + c] = (_Float16)emb[idx * E_ + c];
    }
    __syncthreads();
    for (int nt = wv; nt < G3P / 16; nt += 4) {         // gx = x @ Wih^T (K=64)
      v8f acc = {};
#pragma unroll
      for (int kt = 0; kt < 2; ++kt)
        acc = wmma16(frag_a(sX, EHP, kt * 32, lane), frag_bt(wWihE, 64, nt * 16, kt * 32, lane), acc);
      frag_store(sGx, G3P, nt * 16, acc, lane);
    }
    for (int nt = wv; nt < G3P / 16; nt += 4) {         // gh = h @ Whh^T (K=128)
      v8f acc = {};
#pragma unroll
      for (int kt = 0; kt < 4; ++kt)
        acc = wmma16(frag_a(sHh, HP, kt * 32, lane), frag_bt(wWhhE, HP, nt * 16, kt * 32, lane), acc);
      frag_store(sGh, G3P, nt * 16, acc, lane);
    }
    __syncthreads();
    for (int i = tid; i < BT * H_; i += NTHR) {         // GRU cell
      int r = i / H_, c = i % H_;
      float xr = sGx[r * G3P + c]       + bihE[c];
      float xz = sGx[r * G3P + 100 + c] + bihE[100 + c];
      float xn = sGx[r * G3P + 200 + c] + bihE[200 + c];
      float hr = sGh[r * G3P + c]       + bhhE[c];
      float hz = sGh[r * G3P + 100 + c] + bhhE[100 + c];
      float hn = sGh[r * G3P + 200 + c] + bhhE[200 + c];
      float rg = 1.f / (1.f + __expf(-(xr + hr)));
      float zg = 1.f / (1.f + __expf(-(xz + hz)));
      float ng = tanhf(xn + rg * hn);
      float h2 = (1.f - zg) * ng + zg * sH[r * HP + c];
      sH[r * HP + c]  = h2;
      sHh[r * HP + c] = (_Float16)h2;
      sEnc[(t * BT + r) * H_ + c] = (_Float16)h2;
    }
    __syncthreads();
  }

  // ================= decoder =================
  for (int t = 0; t < S_; ++t) {
    for (int nt = wv; nt < 7; nt += 4) {                // q = h @ attn_W (K=128, N=112)
      v8f acc = {};
#pragma unroll
      for (int kt = 0; kt < 4; ++kt)
        acc = wmma16(frag_a(sHh, HP, kt * 32, lane), frag_bt(wAttnT, HP, nt * 16, kt * 32, lane), acc);
      frag_store(sQ, 112, nt * 16, acc, lane);
    }
    __syncthreads();
    for (int i = tid; i < BT * S_; i += NTHR) {         // score[b,s] = q . enc_out[b,s]
      int r = i % BT, s = i / BT;
      float acc = 0.f;
#pragma unroll 4
      for (int c = 0; c < H_; ++c)
        acc += sQ[r * 112 + c] * (float)sEnc[(s * BT + r) * H_ + c];
      sSc[r * S_ + s] = acc;
    }
    __syncthreads();
    if (tid < BT) {                                     // softmax over S
      int r = tid;
      float m = sSc[r * S_];
#pragma unroll
      for (int s = 1; s < S_; ++s) m = fmaxf(m, sSc[r * S_ + s]);
      float e[S_], sum = 0.f;
#pragma unroll
      for (int s = 0; s < S_; ++s) { e[s] = __expf(sSc[r * S_ + s] - m); sum += e[s]; }
      float inv = 1.f / sum;
#pragma unroll
      for (int s = 0; s < S_; ++s) sSc[r * S_ + s] = e[s] * inv;
    }
    __syncthreads();
    for (int i = tid; i < BT * E_; i += NTHR) {         // x part of decoder input
      int r = i >> 6, c = i & 63;
      int idx = decX[(b0 + r) * S_ + t];
      sX[r * EHP + c] = (_Float16)emb[idx * E_ + c];
    }
    for (int i = tid; i < BT * H_; i += NTHR) {         // ctx part of decoder input
      int r = i / H_, c = i % H_;
      float acc = 0.f;
#pragma unroll
      for (int s = 0; s < S_; ++s)
        acc += sSc[r * S_ + s] * (float)sEnc[(s * BT + r) * H_ + c];
      sX[r * EHP + E_ + c] = (_Float16)acc;
    }
    __syncthreads();
    for (int nt = wv; nt < G3P / 16; nt += 4) {         // gx = [x,ctx] @ Wih^T (K=192)
      v8f acc = {};
#pragma unroll
      for (int kt = 0; kt < 6; ++kt)
        acc = wmma16(frag_a(sX, EHP, kt * 32, lane), frag_bt(wWihD, EHP, nt * 16, kt * 32, lane), acc);
      frag_store(sGx, G3P, nt * 16, acc, lane);
    }
    for (int nt = wv; nt < G3P / 16; nt += 4) {         // gh = h @ Whh^T (K=128)
      v8f acc = {};
#pragma unroll
      for (int kt = 0; kt < 4; ++kt)
        acc = wmma16(frag_a(sHh, HP, kt * 32, lane), frag_bt(wWhhD, HP, nt * 16, kt * 32, lane), acc);
      frag_store(sGh, G3P, nt * 16, acc, lane);
    }
    __syncthreads();
    for (int i = tid; i < BT * H_; i += NTHR) {         // GRU cell (decoder)
      int r = i / H_, c = i % H_;
      float xr = sGx[r * G3P + c]       + bihD[c];
      float xz = sGx[r * G3P + 100 + c] + bihD[100 + c];
      float xn = sGx[r * G3P + 200 + c] + bihD[200 + c];
      float hr = sGh[r * G3P + c]       + bhhD[c];
      float hz = sGh[r * G3P + 100 + c] + bhhD[100 + c];
      float hn = sGh[r * G3P + 200 + c] + bhhD[200 + c];
      float rg = 1.f / (1.f + __expf(-(xr + hr)));
      float zg = 1.f / (1.f + __expf(-(xz + hz)));
      float ng = tanhf(xn + rg * hn);
      float h2 = (1.f - zg) * ng + zg * sH[r * HP + c];
      sH[r * HP + c]  = h2;
      sHh[r * HP + c] = (_Float16)h2;
    }
    __syncthreads();
    for (int nt = wv; nt < 2; nt += 4) {                // logits = h2 @ lin_W^T + b (N=32 pad)
      v8f acc = {};
#pragma unroll
      for (int kt = 0; kt < 4; ++kt)
        acc = wmma16(frag_a(sHh, HP, kt * 32, lane), frag_bt(wLin, HP, nt * 16, kt * 32, lane), acc);
      int n = nt * 16 + (lane & 15), half = lane >> 4;
      if (n < V_) {
        float bb = linb[n];
#pragma unroll
        for (int j = 0; j < 8; ++j) {
          int r = j + half * 8;
          out[((size_t)(b0 + r) * S_ + t) * V_ + n] = acc[j] + bb;
        }
      }
    }
    __syncthreads();
  }
}

extern "C" void kernel_launch(void* const* d_in, const int* in_sizes, int n_in,
                              void* d_out, int out_size, void* d_ws, size_t ws_size,
                              hipStream_t stream) {
  const int*   encX  = (const int*)d_in[0];
  const int*   decX  = (const int*)d_in[1];
  const float* emb   = (const float*)d_in[2];
  const float* WihE  = (const float*)d_in[3];
  const float* WhhE  = (const float*)d_in[4];
  const float* bihE  = (const float*)d_in[5];
  const float* bhhE  = (const float*)d_in[6];
  const float* WihD  = (const float*)d_in[7];
  const float* WhhD  = (const float*)d_in[8];
  const float* bihD  = (const float*)d_in[9];
  const float* bhhD  = (const float*)d_in[10];
  const float* attnW = (const float*)d_in[11];
  const float* linW  = (const float*)d_in[12];
  const float* linb  = (const float*)d_in[13];
  float* out = (float*)d_out;

  _Float16* ws = (_Float16*)d_ws;
  _Float16* wWihE  = ws;
  _Float16* wWhhE  = wWihE + N0;
  _Float16* wWihD  = wWhhE + N1;
  _Float16* wWhhD  = wWihD + N2;
  _Float16* wAttnT = wWhhD + N3;
  _Float16* wLin   = wAttnT + N4;

  prep_weights<<<(NWTOT + 255) / 256, 256, 0, stream>>>(WihE, WhhE, WihD, WhhD, attnW, linW, ws);

  int B = in_sizes[0] / S_;   // encoder_X has B*S elements
  seq2seq_kernel<<<B / BT, NTHR, 0, stream>>>(encX, decX, emb, bihE, bhhE, bihD, bhhD, linb,
                                              wWihE, wWhhE, wWihD, wWhhD, wAttnT, wLin, out);
}
